// HuBERTTransformerEncoder_1520418423184
// MI455X (gfx1250) — compile-verified, matrix operand order checked by reading, not run on hardware
//
#include <hip/hip_runtime.h>
#include <hip/hip_bf16.h>
#include <math.h>

#define DEVI __device__ __forceinline__

namespace {

constexpr int B_  = 4, S_ = 1024, H_ = 768, L_ = 12, NH_ = 12, FF_ = 3072;
constexpr int DH_ = H_ / NH_;   // 64
constexpr int T_  = B_ * S_;    // 4096 tokens
constexpr float EPS_   = 1e-5f;
constexpr float SCALE_ = 0.125f; // 1/sqrt(DH)

typedef _Float16 v8h  __attribute__((ext_vector_type(8)));
typedef _Float16 v16h __attribute__((ext_vector_type(16)));
typedef float    v8f  __attribute__((ext_vector_type(8)));

union Frag16 { v16h v; v8h h[2]; };

DEVI v8f vzero() { v8f z = {0.f,0.f,0.f,0.f,0.f,0.f,0.f,0.f}; return z; }

DEVI v8f wmma_f16(v16h a, v16h b, v8f c) {
  // D = A(16x32 f16) x B(32x16 f16) + C(16x16 f32)
  return __builtin_amdgcn_wmma_f32_16x16x32_f16(false, a, false, b, (short)0, c,
                                                false, false);
}

// CDNA5 16-bit A/B fragment striping (wave32):
//   lanes 0-15  : rows 0-15, elems 0-7 -> K = 0..7,  elems 8-15 -> K = 16..23
//   lanes 16-31 : rows 0-15, elems 0-7 -> K = 8..15, elems 8-15 -> K = 24..31
DEVI v16h load_frag(const _Float16* base, int stride, int row_base, int k0, int lane) {
  const int r  = row_base + (lane & 15);
  const int kh = (lane >> 4) << 3;  // 0 or 8
  const _Float16* p = base + (size_t)r * stride + k0 + kh;
  Frag16 f;
  f.h[0] = *(const v8h*)(p);
  f.h[1] = *(const v8h*)(p + 16);
  return f.v;
}

DEVI int   f2i(float f) { union { float f; int i; } u; u.f = f; return u.i; }
DEVI float i2f(int i)   { union { int i; float f; } u; u.i = i; return u.f; }

template<int XM>
DEVI float swz_xor(float x) {
  // ds_swizzle group-of-32: offset = {xor[14:10], or[9:5], and[4:0]}
  return i2f(__builtin_amdgcn_ds_swizzle(f2i(x), (XM << 10) | 0x1f));
}
DEVI float half_max(float v) {      // reduce across a 16-lane half-group
  v = fmaxf(v, swz_xor<1>(v)); v = fmaxf(v, swz_xor<2>(v));
  v = fmaxf(v, swz_xor<4>(v)); v = fmaxf(v, swz_xor<8>(v));
  return v;
}
DEVI float half_sum(float v) {
  v += swz_xor<1>(v); v += swz_xor<2>(v);
  v += swz_xor<4>(v); v += swz_xor<8>(v);
  return v;
}

// ---------------- one-time weight transpose + fp32 -> f16 ----------------
// src [layer][K][N] f32  ->  dst [layer][N][K] f16   (grid.z = layer)
__global__ __launch_bounds__(256)
void tr_cvt_kernel(const float* __restrict__ src, _Float16* __restrict__ dst,
                   int K, int N) {
  __shared__ float tile[32][33];
  const float* s = src + (size_t)blockIdx.z * K * N;
  _Float16*    d = dst + (size_t)blockIdx.z * K * N;
  const int kb = blockIdx.y * 32, nb = blockIdx.x * 32;
  const int tx = threadIdx.x & 31, ty = threadIdx.x >> 5;  // 32x8
  for (int i = ty; i < 32; i += 8) tile[i][tx] = s[(size_t)(kb + i) * N + nb + tx];
  __syncthreads();
  for (int i = ty; i < 32; i += 8)
    d[(size_t)(nb + i) * K + kb + tx] = (_Float16)tile[tx][i];
}

// ---------------- elementwise ----------------
__global__ void addpos_kernel(const float* __restrict__ h, const float* __restrict__ pos,
                              float* __restrict__ o) {
  size_t i = (size_t)blockIdx.x * 256 + threadIdx.x;
  if (i >= (size_t)T_ * H_) return;
  const int c = (int)(i % H_);
  const int s = (int)((i / H_) % S_);
  o[i] = h[i] + pos[(size_t)s * H_ + c];
}

// ---------------- LayerNorm (block per token) ----------------
template<bool F16OUT>
__global__ __launch_bounds__(256)
void ln_kernel(const float* __restrict__ x, const float* __restrict__ sc,
               const float* __restrict__ bi, void* __restrict__ outp) {
  __shared__ float sh[256];
  const int t = blockIdx.x;
  const float* xp = x + (size_t)t * H_;
  const int tid = threadIdx.x;

  float v = 0.f;
  for (int i = tid; i < H_; i += 256) v += xp[i];
  sh[tid] = v; __syncthreads();
  for (int o = 128; o > 0; o >>= 1) { if (tid < o) sh[tid] += sh[tid + o]; __syncthreads(); }
  const float mean = sh[0] / H_;
  __syncthreads();

  float v2 = 0.f;
  for (int i = tid; i < H_; i += 256) { float d = xp[i] - mean; v2 += d * d; }
  sh[tid] = v2; __syncthreads();
  for (int o = 128; o > 0; o >>= 1) { if (tid < o) sh[tid] += sh[tid + o]; __syncthreads(); }
  const float rstd = rsqrtf(sh[0] / H_ + EPS_);

  for (int i = tid; i < H_; i += 256) {
    const float y = (xp[i] - mean) * rstd * sc[i] + bi[i];
    if (F16OUT) ((_Float16*)outp)[(size_t)t * H_ + i] = (_Float16)y;
    else        ((float*)outp)[(size_t)t * H_ + i]    = y;
  }
}

// ---------------- WMMA GEMM ----------------
// y = act( A[M x K] @ Wt[N x K]^T + bias[N] ), f16 inputs, f32 accumulate.
// Wave tile 32x64 (8 WMMA per K-step, 12 b128 loads -> 1.5 loads/WMMA).
// Block = 4 waves = 64x128 tile.
// MODE 0: f16 plain row-major [row][col]
// MODE 1: f16 Q/K layout  [b][head][s][dh]
// MODE 2: f16 V transposed [b][head][dh][s]
// MODE 3: f32 residual accumulate: out[row*N+col] += y   (fused h = h + FFN(y))
template<int MODE, bool GELU>
__global__ __launch_bounds__(128)
void gemm_kernel(const _Float16* __restrict__ A, const _Float16* __restrict__ Wt,
                 const float* __restrict__ bias, void* __restrict__ outp,
                 int K, int N) {
  const int lane = threadIdx.x & 31;
  const int wave = threadIdx.x >> 5;               // 0..3
  const int m0 = blockIdx.y * 64  + (wave >> 1) * 32;
  const int n0 = blockIdx.x * 128 + (wave & 1) * 64;

  v8f acc[2][4];
  #pragma unroll
  for (int i = 0; i < 2; i++)
    #pragma unroll
    for (int j = 0; j < 4; j++) acc[i][j] = vzero();

  for (int k0 = 0; k0 < K; k0 += 32) {
    __builtin_prefetch(Wt + (size_t)(n0 + (lane & 15)) * K + k0 + 128, 0, 1);
    const v16h a0 = load_frag(A, K, m0,      k0, lane);
    const v16h a1 = load_frag(A, K, m0 + 16, k0, lane);
    #pragma unroll
    for (int j = 0; j < 4; j++) {
      const v16h bj = load_frag(Wt, K, n0 + j * 16, k0, lane);
      acc[0][j] = wmma_f16(a0, bj, acc[0][j]);
      acc[1][j] = wmma_f16(a1, bj, acc[1][j]);
    }
  }

  #pragma unroll
  for (int i = 0; i < 2; i++)
  #pragma unroll
  for (int j = 0; j < 4; j++) {
    const int col = n0 + j * 16 + (lane & 15);
    const float bj = bias[col];
    #pragma unroll
    for (int r = 0; r < 8; r++) {
      const int row = m0 + i * 16 + r + ((lane >> 4) << 3);
      float v = acc[i][j][r] + bj;
      if (GELU) v = 0.5f * v * (1.f + erff(v * 0.70710678118f));
      if (MODE == 3) {
        ((float*)outp)[(size_t)row * N + col] += v;
      } else if (MODE == 0) {
        ((_Float16*)outp)[(size_t)row * N + col] = (_Float16)v;
      } else {
        const int b  = row >> 10;            // /S_
        const int s  = row & (S_ - 1);
        const int hd = col / DH_;
        const int dh = col & (DH_ - 1);
        size_t idx;
        if (MODE == 1) idx = ((size_t)(b * NH_ + hd) * S_  + s)  * DH_ + dh;
        else           idx = ((size_t)(b * NH_ + hd) * DH_ + dh) * S_  + s;
        ((_Float16*)outp)[idx] = (_Float16)v;
      }
    }
  }
}

// ---------------- fused attention (flash-style, online softmax) ----------------
// Q,K: [b][head][s][dh] f16 ; Vt: [b][head][dh][s] f16
// Epilogue fuses the residual: hres[b][s][hd*DH+dh] += ctx  (f32, one writer/elem)
__global__ __launch_bounds__(128)
void attn_kernel(const _Float16* __restrict__ Q, const _Float16* __restrict__ Kk,
                 const _Float16* __restrict__ Vt, const int* __restrict__ mask,
                 float* __restrict__ hres) {
  __shared__ _Float16 plds[4][16][40];   // per-wave P-tile transpose buffer
  const int lane = threadIdx.x & 31;
  const int wave = threadIdx.x >> 5;
  const int bh = blockIdx.x;             // b*NH + head
  const int b  = bh / NH_;
  const int hd = bh % NH_;
  const int q0 = blockIdx.y * 64 + wave * 16;

  const _Float16* Qb = Q  + (size_t)bh * S_ * DH_;
  const _Float16* Kb = Kk + (size_t)bh * S_ * DH_;
  const _Float16* Vb = Vt + (size_t)bh * DH_ * S_;
  const int* mb = mask + b * S_;

  const v16h qA0 = load_frag(Qb, DH_, q0, 0,  lane);
  const v16h qA1 = load_frag(Qb, DH_, q0, 32, lane);

  v8f acc[4];
  #pragma unroll
  for (int j = 0; j < 4; j++) acc[j] = vzero();
  float mrow[8], lrow[8];
  #pragma unroll
  for (int r = 0; r < 8; r++) { mrow[r] = -1e30f; lrow[r] = 0.f; }

  for (int kb = 0; kb < S_; kb += 32) {
    // ---- scores: S[16q x 32k] = Q(16x64) @ K^T(64x32) ----
    v8f s0 = vzero(), s1 = vzero();
    {
      v16h kf;
      kf = load_frag(Kb, DH_, kb,      0,  lane); s0 = wmma_f16(qA0, kf, s0);
      kf = load_frag(Kb, DH_, kb,      32, lane); s0 = wmma_f16(qA1, kf, s0);
      kf = load_frag(Kb, DH_, kb + 16, 0,  lane); s1 = wmma_f16(qA0, kf, s1);
      kf = load_frag(Kb, DH_, kb + 16, 32, lane); s1 = wmma_f16(qA1, kf, s1);
    }
    const float mb0 = (mb[kb + (lane & 15)]      == 0) ? -10000.f : 0.f;
    const float mb1 = (mb[kb + 16 + (lane & 15)] == 0) ? -10000.f : 0.f;

    float rm[8];
    #pragma unroll
    for (int r = 0; r < 8; r++) {
      s0[r] = s0[r] * SCALE_ + mb0;
      s1[r] = s1[r] * SCALE_ + mb1;
      rm[r] = fmaxf(s0[r], s1[r]);
    }
    #pragma unroll
    for (int r = 0; r < 8; r++) rm[r] = half_max(rm[r]);

    float corr[8], rs[8];
    #pragma unroll
    for (int r = 0; r < 8; r++) {
      const float mn = fmaxf(mrow[r], rm[r]);
      corr[r] = __expf(mrow[r] - mn);
      mrow[r] = mn;
      s0[r] = __expf(s0[r] - mn);
      s1[r] = __expf(s1[r] - mn);
      rs[r] = s0[r] + s1[r];
    }
    #pragma unroll
    for (int r = 0; r < 8; r++) rs[r] = half_sum(rs[r]);
    #pragma unroll
    for (int r = 0; r < 8; r++) lrow[r] = lrow[r] * corr[r] + rs[r];
    #pragma unroll
    for (int j = 0; j < 4; j++)
      #pragma unroll
      for (int r = 0; r < 8; r++) acc[j][r] *= corr[r];

    // ---- transpose P from C-layout to A-layout through wave-private LDS ----
    #pragma unroll
    for (int r = 0; r < 8; r++) {
      const int row = r + ((lane >> 4) << 3);
      plds[wave][row][lane & 15]        = (_Float16)s0[r];
      plds[wave][row][16 + (lane & 15)] = (_Float16)s1[r];
    }
    asm volatile("s_wait_dscnt 0" ::: "memory");  // wave-private slab: no barrier
    const v16h pf = load_frag(&plds[wave][0][0], 40, 0, 0, lane);

    // ---- ctx += P(16x32) @ V(32x64) ----
    #pragma unroll
    for (int j = 0; j < 4; j++) {
      v16h vf = load_frag(Vb, S_, j * 16, kb, lane);
      acc[j] = wmma_f16(pf, vf, acc[j]);
    }
  }

  // normalize and fuse residual add into the f32 hidden state
  #pragma unroll
  for (int j = 0; j < 4; j++)
    #pragma unroll
    for (int r = 0; r < 8; r++) {
      const int row = q0 + r + ((lane >> 4) << 3);
      const int dh  = j * 16 + (lane & 15);
      const float v = acc[j][r] / lrow[r];
      hres[((size_t)(b * S_ + row)) * H_ + hd * DH_ + dh] += v;
    }
}

} // anonymous namespace

// ---------------- host orchestration ----------------
extern "C" void kernel_launch(void* const* d_in, const int* in_sizes, int n_in,
                              void* d_out, int out_size, void* d_ws, size_t ws_size,
                              hipStream_t stream) {
  (void)in_sizes; (void)n_in; (void)out_size; (void)ws_size;
  const float* hidden = (const float*)d_in[0];
  const int*   amask  = (const int*)d_in[1];
  const float* pos    = (const float*)d_in[2];
  const float* Wq   = (const float*)d_in[3];
  const float* bq   = (const float*)d_in[4];
  const float* Wk   = (const float*)d_in[5];
  const float* bk   = (const float*)d_in[6];
  const float* Wv   = (const float*)d_in[7];
  const float* bv   = (const float*)d_in[8];
  const float* ln1s = (const float*)d_in[9];
  const float* ln1b = (const float*)d_in[10];
  const float* ln2s = (const float*)d_in[11];
  const float* ln2b = (const float*)d_in[12];
  const float* W1   = (const float*)d_in[13];
  const float* b1   = (const float*)d_in[14];
  const float* W2   = (const float*)d_in[15];
  const float* b2   = (const float*)d_in[16];
  const float* lnfs = (const float*)d_in[17];
  const float* lnfb = (const float*)d_in[18];

  char* wsp = (char*)d_ws;
  auto take = [&](size_t bytes) -> char* {
    char* p = wsp;
    wsp += (bytes + 255) & ~(size_t)255;
    return p;
  };
  _Float16* wq_t = (_Float16*)take((size_t)L_ * H_ * H_ * 2);
  _Float16* wk_t = (_Float16*)take((size_t)L_ * H_ * H_ * 2);
  _Float16* wv_t = (_Float16*)take((size_t)L_ * H_ * H_ * 2);
  _Float16* w1_t = (_Float16*)take((size_t)L_ * H_ * FF_ * 2);
  _Float16* w2_t = (_Float16*)take((size_t)L_ * H_ * FF_ * 2);
  float*    hbuf = (float*)   take((size_t)T_ * H_ * 4);
  _Float16* xf16 = (_Float16*)take((size_t)T_ * H_ * 2);
  _Float16* qf   = (_Float16*)take((size_t)T_ * H_ * 2);
  _Float16* kf   = (_Float16*)take((size_t)T_ * H_ * 2);
  _Float16* vf   = (_Float16*)take((size_t)T_ * H_ * 2);
  _Float16* midb = (_Float16*)take((size_t)T_ * FF_ * 2);

  // one-time weight convert+transpose (f32 [K][N] -> f16 [N][K])
  tr_cvt_kernel<<<dim3(H_ / 32,  H_ / 32,  L_), 256, 0, stream>>>(Wq, wq_t, H_,  H_);
  tr_cvt_kernel<<<dim3(H_ / 32,  H_ / 32,  L_), 256, 0, stream>>>(Wk, wk_t, H_,  H_);
  tr_cvt_kernel<<<dim3(H_ / 32,  H_ / 32,  L_), 256, 0, stream>>>(Wv, wv_t, H_,  H_);
  tr_cvt_kernel<<<dim3(FF_ / 32, H_ / 32,  L_), 256, 0, stream>>>(W1, w1_t, H_,  FF_);
  tr_cvt_kernel<<<dim3(H_ / 32,  FF_ / 32, L_), 256, 0, stream>>>(W2, w2_t, FF_, H_);

  const int nElem = (int)(((size_t)T_ * H_ + 255) / 256);
  addpos_kernel<<<nElem, 256, 0, stream>>>(hidden, pos, hbuf);

  const dim3 gH (H_ / 128,  T_ / 64);  // N=768 GEMMs
  const dim3 gFF(FF_ / 128, T_ / 64);  // N=3072 GEMM
  const dim3 gAtt(B_ * NH_, S_ / 64);

  for (int l = 0; l < L_; ++l) {
    const size_t wOff  = (size_t)l * H_ * H_;
    const size_t w1Off = (size_t)l * H_ * FF_;

    ln_kernel<true><<<T_, 256, 0, stream>>>(hbuf, ln1s + l * H_, ln1b + l * H_, xf16);

    gemm_kernel<1, false><<<gH, 128, 0, stream>>>(xf16, wq_t + wOff, bq + l * H_, qf, H_, H_);
    gemm_kernel<1, false><<<gH, 128, 0, stream>>>(xf16, wk_t + wOff, bk + l * H_, kf, H_, H_);
    gemm_kernel<2, false><<<gH, 128, 0, stream>>>(xf16, wv_t + wOff, bv + l * H_, vf, H_, H_);

    // attention with fused "h += ctx"
    attn_kernel<<<gAtt, 128, 0, stream>>>(qf, kf, vf, amask, hbuf);

    ln_kernel<true><<<T_, 256, 0, stream>>>(hbuf, ln2s + l * H_, ln2b + l * H_, xf16);

    gemm_kernel<0, true ><<<gFF, 128, 0, stream>>>(xf16, w1_t + w1Off, b1 + l * FF_, midb, H_,  FF_);
    // FFN-2 with fused "h += y"
    gemm_kernel<3, false><<<gH,  128, 0, stream>>>(midb, w2_t + w1Off, b2 + l * H_,  hbuf, FF_, H_);
  }

  ln_kernel<false><<<T_, 256, 0, stream>>>(hbuf, lnfs, lnfb, d_out);
}
